// Mamba2Mixer_9895604650104
// MI455X (gfx1250) — compile-verified, matrix operand order checked by reading, not run on hardware
//
#include <hip/hip_runtime.h>
#include <hip/hip_bf16.h>
#include <math.h>

#define B_     2
#define S_     4096
#define HSZ_   1024
#define INTER_ 2048
#define NH_    32
#define HD_    64
#define NST_   128
#define KC_    4
#define CHK_   256
#define NC_    16
#define PROJ_  4384
#define CONVD_ 2304
#define EPSF   1e-5f

typedef __attribute__((ext_vector_type(16))) __bf16 v16bf;
typedef __attribute__((ext_vector_type(8)))  float  v8f;
typedef __attribute__((ext_vector_type(4)))  float  v4f;
typedef __attribute__((ext_vector_type(4)))  unsigned int u32x4;
typedef __attribute__((ext_vector_type(8)))  int  i32x8;
typedef __attribute__((ext_vector_type(4)))  int  i32x4;

static __device__ inline unsigned short f2bfu(float f) {
  union { float f; unsigned u; } v; v.f = f;
  unsigned r = v.u + 0x7FFFu + ((v.u >> 16) & 1u);
  return (unsigned short)(r >> 16);
}

static __device__ inline float siluf(float x) { return x / (1.f + __expf(-x)); }

// Load 16 contiguous f32, round to bf16, return WMMA fragment.
static __device__ inline v16bf ldfrag_cvt(const float* __restrict__ p) {
  union { v16bf v; unsigned short s[16]; } r;
#pragma unroll
  for (int i = 0; i < 16; i += 4) {
    v4f x = *(const v4f*)(p + i);
    r.s[i + 0] = f2bfu(x[0]);
    r.s[i + 1] = f2bfu(x[1]);
    r.s[i + 2] = f2bfu(x[2]);
    r.s[i + 3] = f2bfu(x[3]);
  }
  return r.v;
}

static __device__ inline v8f wmma_bf16(v16bf a, v16bf b, v8f c) {
  return __builtin_amdgcn_wmma_f32_16x16x32_bf16(false, a, false, b, (short)0, c,
                                                 false, false);
}

// TDM: DMA `nelem` contiguous 2-byte elements from global into LDS.
// Descriptor per CDNA5 ISA 8.3/8.4 (group0: flags/lds/global/type,
// group1: data_size + dims/strides). 6-arg builtin on this toolchain.
static __device__ inline void tdm_load_lds_b16(unsigned lds_off,
                                               const void* gptr,
                                               unsigned nelem) {
  const unsigned long long ga = (unsigned long long)gptr;
  u32x4 g0;
  g0[0] = 1u;                                   // count=1, user mode
  g0[1] = lds_off;                              // lds_addr (bytes)
  g0[2] = (unsigned)(ga & 0xFFFFFFFFull);       // global_addr[31:0]
  g0[3] = (unsigned)((ga >> 32) & 0x01FFFFFFull) | (2u << 30);  // addr hi, type=2
  i32x8 g1;
  g1[0] = (int)(1u << 16);                      // wg_mask=0, data_size=1 (2B)
  g1[1] = (int)((nelem & 0xFFFFu) << 16);       // tensor_dim0[15:0]
  g1[2] = (int)((nelem >> 16) | (1u << 16));    // tensor_dim0[31:16], tensor_dim1=1
  g1[3] = (int)((nelem & 0xFFFFu) << 16);       // tile_dim0 = nelem (<=65535)
  g1[4] = 1;                                    // tile_dim1=1, tile_dim2=0
  g1[5] = (int)nelem;                           // tensor_dim0_stride[31:0]
  g1[6] = 0;                                    // stride0[47:32], stride1 lo
  g1[7] = 0;
  i32x4 g2 = {};
  i32x4 g3 = {};
  i32x8 g4 = {};
  __builtin_amdgcn_tensor_load_to_lds(g0, g1, g2, g3, g4, 0);
}

// ---------------------------------------------------------------- converters
__global__ void k_cvt_bf16(const float* __restrict__ in,
                           unsigned short* __restrict__ out, long n) {
  long i = (long)blockIdx.x * blockDim.x + threadIdx.x;
  long stride = (long)gridDim.x * blockDim.x;
  for (; i < n; i += stride) out[i] = f2bfu(in[i]);
}

// ------------------------------------------------------- generic bf16 GEMM
// C[M,N] = A[M,K] * W[N,K]^T   (A, W bf16 row-major over K; C f32)
// block 128x128, 8 waves, wave tile 64x32 (4x2 WMMA tiles)
__global__ __launch_bounds__(256) void k_gemm_bf16(
    const unsigned short* __restrict__ A, const unsigned short* __restrict__ W,
    float* __restrict__ Cout, int M, int N, int K) {
  const int wv = threadIdx.x >> 5;
  const int lane = threadIdx.x & 31;
  const int l15 = lane & 15;
  const int kh = (lane >> 4) * 16;
  const int wm = blockIdx.y * 128 + (wv >> 2) * 64;
  const int wn = blockIdx.x * 128 + (wv & 3) * 32;
  v8f acc[4][2] = {};

  for (int k0 = 0; k0 < K; k0 += 32) {
    if (k0 + 32 < K) {
      __builtin_prefetch(A + (size_t)(wm + l15) * K + k0 + 32 + kh, 0, 3);
      __builtin_prefetch(W + (size_t)(wn + l15) * K + k0 + 32 + kh, 0, 3);
    }
    v16bf af[4], bf[2];
#pragma unroll
    for (int i = 0; i < 4; ++i)
      af[i] = *(const v16bf*)(A + (size_t)(wm + i * 16 + l15) * K + k0 + kh);
#pragma unroll
    for (int j = 0; j < 2; ++j)
      bf[j] = *(const v16bf*)(W + (size_t)(wn + j * 16 + l15) * K + k0 + kh);
#pragma unroll
    for (int i = 0; i < 4; ++i)
#pragma unroll
      for (int j = 0; j < 2; ++j) acc[i][j] = wmma_bf16(af[i], bf[j], acc[i][j]);
  }

  const int rhi = (lane >> 4) * 8;
#pragma unroll
  for (int i = 0; i < 4; ++i)
#pragma unroll
    for (int j = 0; j < 2; ++j) {
      const int col = wn + j * 16 + l15;
      if (col < N) {
#pragma unroll
        for (int e = 0; e < 8; ++e) {
          const int row = wm + i * 16 + rhi + e;
          Cout[(size_t)row * N + col] = acc[i][j][e];
        }
      }
    }
}

// ------------------------------------------- depthwise conv + SiLU + softplus
__global__ __launch_bounds__(256) void k_conv_act(
    const float* __restrict__ proj, const float* __restrict__ convw,
    const float* __restrict__ convb, const float* __restrict__ dtb,
    float* __restrict__ hbuf, float* __restrict__ Bbuf,
    float* __restrict__ Cbuf, float* __restrict__ dtbuf) {
  const int row = blockIdx.x;  // b*S + s
  const int s = row & (S_ - 1);
  for (int ch = threadIdx.x; ch < CONVD_ + NH_; ch += 256) {
    if (ch < CONVD_) {
      float acc = convb[ch];
#pragma unroll
      for (int j = 0; j < KC_; ++j) {
        const int sp = s - (KC_ - 1) + j;
        if (sp >= 0)
          acc += proj[(size_t)(row - (KC_ - 1) + j) * PROJ_ + INTER_ + ch] *
                 convw[ch * KC_ + j];
      }
      acc = siluf(acc);
      if (ch < INTER_)
        hbuf[(size_t)row * INTER_ + ch] = acc;
      else if (ch < INTER_ + NST_)
        Bbuf[(size_t)row * NST_ + (ch - INTER_)] = acc;
      else
        Cbuf[(size_t)row * NST_ + (ch - INTER_ - NST_)] = acc;
    } else {
      const int hh = ch - CONVD_;
      const float x = proj[(size_t)row * PROJ_ + INTER_ + CONVD_ + hh] + dtb[hh];
      const float sp = (x > 20.f) ? x : log1pf(__expf(x));
      dtbuf[(size_t)row * NH_ + hh] = fmaxf(sp, 0.f);
    }
  }
}

// ---------------------------- per-(b,c,h): cumsum(a), g, states = xdt^T @ B'
__global__ __launch_bounds__(256) void k_chunk_state(
    const float* __restrict__ hbuf, const float* __restrict__ Bbuf,
    const float* __restrict__ dtbuf, const float* __restrict__ A_log,
    float* __restrict__ cAg, float* __restrict__ gws,
    float* __restrict__ states) {
  __shared__ float s_cA[CHK_];
  __shared__ float s_dt[CHK_];
  __shared__ unsigned short sBT[NST_ * 136];  // [n][l-half], stride 136
  __shared__ unsigned short sXT[HD_ * 136];   // [p][l-half]
  const int bid = blockIdx.x;
  const int h = bid % NH_;
  const int c = (bid / NH_) % NC_;
  const int b = bid / (NH_ * NC_);
  const int tid = threadIdx.x;
  const size_t rowbase = (size_t)(b * S_ + c * CHK_);

  const float Aneg = -__expf(A_log[h]);
  const float dtv = dtbuf[(rowbase + tid) * NH_ + h];
  s_dt[tid] = dtv;
  s_cA[tid] = dtv * Aneg;
  __syncthreads();
  for (int off = 1; off < CHK_; off <<= 1) {  // inclusive scan
    const float add = (tid >= off) ? s_cA[tid - off] : 0.f;
    __syncthreads();
    s_cA[tid] += add;
    __syncthreads();
  }
  const float cAl = s_cA[tid];
  cAg[(size_t)bid * CHK_ + tid] = cAl;
  const float tot = s_cA[CHK_ - 1];
  if (tid == 0) gws[(b * NH_ + h) * NC_ + c] = __expf(tot);

  const int wv = tid >> 5, lane = tid & 31, l15 = lane & 15, kh = (lane >> 4) * 16;
  const int pt = wv >> 1;        // p-tile 0..3
  const int nt0 = (wv & 1) * 4;  // n-tile base
  v8f acc[4] = {};

  for (int half = 0; half < 2; ++half) {
    const int lb = half * 128;
    __syncthreads();
    for (int idx = tid; idx < 128 * NST_; idx += 256) {  // B^T with decay
      const int lr = idx >> 7, n = idx & 127;
      const int l = lb + lr;
      const float d = __expf(tot - s_cA[l]);
      sBT[n * 136 + lr] = f2bfu(Bbuf[(rowbase + l) * NST_ + n] * d);
    }
    for (int idx = tid; idx < 128 * HD_; idx += 256) {  // xdt^T
      const int lr = idx >> 6, p = idx & 63;
      const int l = lb + lr;
      sXT[p * 136 + lr] =
          f2bfu(hbuf[(rowbase + l) * INTER_ + h * HD_ + p] * s_dt[l]);
    }
    __syncthreads();
#pragma unroll
    for (int ks = 0; ks < 4; ++ks) {
      const int ko = ks * 32 + kh;
      const v16bf a = *(const v16bf*)(&sXT[(pt * 16 + l15) * 136 + ko]);
#pragma unroll
      for (int j = 0; j < 4; ++j) {
        const v16bf bb = *(const v16bf*)(&sBT[((nt0 + j) * 16 + l15) * 136 + ko]);
        acc[j] = wmma_bf16(a, bb, acc[j]);
      }
    }
  }
  const size_t sbase = ((size_t)(b * NH_ + h) * NC_ + c) * (HD_ * NST_);
  const int rhi = (lane >> 4) * 8;
#pragma unroll
  for (int j = 0; j < 4; ++j) {
    const int n = (nt0 + j) * 16 + l15;
#pragma unroll
    for (int e = 0; e < 8; ++e)
      states[sbase + (size_t)(pt * 16 + rhi + e) * NST_ + n] = acc[j][e];
  }
}

// ------------------- inter-chunk scan; emits running "prev" states in bf16
__global__ __launch_bounds__(256) void k_scan(
    const float* __restrict__ states, const float* __restrict__ gws,
    unsigned short* __restrict__ prevbf) {
  const int bh = blockIdx.x;  // b*NH + h
  const int tid = threadIdx.x;
  float carry[32];
#pragma unroll
  for (int k = 0; k < 32; ++k) carry[k] = 0.f;
  for (int c = 0; c < NC_; ++c) {
    const float gc = gws[bh * NC_ + c];
    const size_t base = ((size_t)bh * NC_ + c) * (HD_ * NST_);
#pragma unroll
    for (int k = 0; k < 32; ++k) {
      const size_t idx = base + (size_t)k * 256 + tid;
      prevbf[idx] = f2bfu(carry[k]);
      carry[k] = carry[k] * gc + states[idx];
    }
  }
}

// ---------------------------------- G = mask(L) o (C @ B^T), stored bf16
__global__ __launch_bounds__(256) void k_chunk_G(
    const float* __restrict__ Bbuf, const float* __restrict__ Cbuf,
    const float* __restrict__ cAg, unsigned short* __restrict__ G) {
  __shared__ float s_cA[CHK_];
  const int panel = blockIdx.x & 3;
  const int bid = blockIdx.x >> 2;
  const int c = (bid / NH_) % NC_;
  const int b = bid / (NH_ * NC_);
  const int tid = threadIdx.x;
  s_cA[tid] = cAg[(size_t)bid * CHK_ + tid];
  __syncthreads();
  const int wv = tid >> 5, lane = tid & 31, l15 = lane & 15, kh = (lane >> 4) * 16;
  const int lt = panel * 64 + (wv >> 1) * 16;
  const int st0 = (wv & 1) * 8;
  const size_t rowbase = (size_t)(b * S_ + c * CHK_);
  v8f acc[8] = {};
#pragma unroll
  for (int ks = 0; ks < 4; ++ks) {
    const int ko = ks * 32 + kh;
    const v16bf a = ldfrag_cvt(&Cbuf[(rowbase + lt + l15) * NST_ + ko]);
#pragma unroll
    for (int j = 0; j < 8; ++j) {
      const v16bf bb =
          ldfrag_cvt(&Bbuf[(rowbase + (st0 + j) * 16 + l15) * NST_ + ko]);
      acc[j] = wmma_bf16(a, bb, acc[j]);
    }
  }
  unsigned short* Gp = G + (size_t)bid * (CHK_ * CHK_);
  const int rhi = (lane >> 4) * 8;
#pragma unroll
  for (int j = 0; j < 8; ++j) {
    const int s = (st0 + j) * 16 + l15;
#pragma unroll
    for (int e = 0; e < 8; ++e) {
      const int l = lt + rhi + e;
      const float v = (l >= s) ? acc[j][e] * __expf(s_cA[l] - s_cA[s]) : 0.f;
      Gp[(size_t)l * CHK_ + s] = f2bfu(v);
    }
  }
}

// --------------- Y = exp(cA) o (C @ prev^T) + G @ xdt + D*x   (y aliases x)
// prev tile (bf16, 16KB) is staged into LDS by the Tensor Data Mover.
__global__ __launch_bounds__(256) void k_chunk_out(
    const float* __restrict__ Cbuf, const unsigned short* __restrict__ G,
    const unsigned short* __restrict__ prevbf, const float* __restrict__ dtbuf,
    const float* __restrict__ cAg, const float* __restrict__ Dv, float* xy) {
  __shared__ float s_cA[CHK_];
  __shared__ unsigned short sXT[HD_ * 264];    // xdt^T, stride 264
  __shared__ unsigned short sPrev[HD_ * NST_]; // prev tile (bf16), TDM target
  const int bid = blockIdx.x;
  const int h = bid % NH_;
  const int c = (bid / NH_) % NC_;
  const int b = bid / (NH_ * NC_);
  const int tid = threadIdx.x;
  const size_t rowbase = (size_t)(b * S_ + c * CHK_);
  const size_t pbase = ((size_t)(b * NH_ + h) * NC_ + c) * (HD_ * NST_);

  if (tid < 32) {  // wave 0 issues the tensor DMA (EXEC ignored by TDM)
    tdm_load_lds_b16((unsigned)(unsigned long long)(const void*)&sPrev[0],
                     prevbf + pbase, HD_ * NST_);
  }
  s_cA[tid] = cAg[(size_t)bid * CHK_ + tid];
  for (int idx = tid; idx < CHK_ * HD_; idx += 256) {
    const int l = idx >> 6, p = idx & 63;
    const float v = xy[(rowbase + l) * INTER_ + h * HD_ + p] *
                    dtbuf[(rowbase + l) * NH_ + h];
    sXT[p * 264 + l] = f2bfu(v);
  }
  __builtin_amdgcn_s_wait_tensorcnt(0);
  __syncthreads();

  const int wv = tid >> 5, lane = tid & 31, l15 = lane & 15, kh = (lane >> 4) * 16;
  v8f acc[2][4] = {};

  // Yo: K = 128 over n; B-frags read bf16 straight from TDM-staged LDS
#pragma unroll
  for (int ks = 0; ks < 4; ++ks) {
    const int ko = ks * 32 + kh;
    const v16bf a0 =
        ldfrag_cvt(&Cbuf[(rowbase + (wv * 2 + 0) * 16 + l15) * NST_ + ko]);
    const v16bf a1 =
        ldfrag_cvt(&Cbuf[(rowbase + (wv * 2 + 1) * 16 + l15) * NST_ + ko]);
#pragma unroll
    for (int jp = 0; jp < 4; ++jp) {
      const v16bf bb = *(const v16bf*)(&sPrev[(jp * 16 + l15) * NST_ + ko]);
      acc[0][jp] = wmma_bf16(a0, bb, acc[0][jp]);
      acc[1][jp] = wmma_bf16(a1, bb, acc[1][jp]);
    }
  }
  // scale rows by exp(cA[l])
  const int rhi = (lane >> 4) * 8;
#pragma unroll
  for (int i = 0; i < 2; ++i)
#pragma unroll
    for (int e = 0; e < 8; ++e) {
      const float sc = __expf(s_cA[(wv * 2 + i) * 16 + rhi + e]);
#pragma unroll
      for (int jp = 0; jp < 4; ++jp) acc[i][jp][e] *= sc;
    }
  // Yd: K = 256 over s; A-frags straight from bf16 G in global
  const unsigned short* Gp = G + (size_t)bid * (CHK_ * CHK_);
#pragma unroll
  for (int ks = 0; ks < 8; ++ks) {
    const int ko = ks * 32 + kh;
    const v16bf a0 =
        *(const v16bf*)(Gp + (size_t)((wv * 2 + 0) * 16 + l15) * CHK_ + ko);
    const v16bf a1 =
        *(const v16bf*)(Gp + (size_t)((wv * 2 + 1) * 16 + l15) * CHK_ + ko);
#pragma unroll
    for (int jp = 0; jp < 4; ++jp) {
      const v16bf bb = *(const v16bf*)(&sXT[(jp * 16 + l15) * 264 + ko]);
      acc[0][jp] = wmma_bf16(a0, bb, acc[0][jp]);
      acc[1][jp] = wmma_bf16(a1, bb, acc[1][jp]);
    }
  }
  // add D*x and store (y aliases x; each lane reads/writes its own element)
  const float Dh = Dv[h];
#pragma unroll
  for (int i = 0; i < 2; ++i)
#pragma unroll
    for (int jp = 0; jp < 4; ++jp) {
      const int p = jp * 16 + l15;
#pragma unroll
      for (int e = 0; e < 8; ++e) {
        const int l = (wv * 2 + i) * 16 + rhi + e;
        const size_t xi = (rowbase + l) * INTER_ + h * HD_ + p;
        xy[xi] = acc[i][jp][e] + Dh * xy[xi];
      }
    }
}

// ----------------------------------------- gated RMSNorm -> bf16 activations
__global__ __launch_bounds__(256) void k_norm(const float* __restrict__ y,
                                              const float* __restrict__ proj,
                                              const float* __restrict__ nw,
                                              unsigned short* __restrict__ out) {
  __shared__ float s_red[256];
  const int row = blockIdx.x;
  const int tid = threadIdx.x;
  float yf[8];
  float ss = 0.f;
#pragma unroll
  for (int k = 0; k < 8; ++k) {
    const int i = tid + k * 256;
    const float g = proj[(size_t)row * PROJ_ + i];
    const float v = y[(size_t)row * INTER_ + i] * siluf(g);
    yf[k] = v;
    ss += v * v;
  }
  s_red[tid] = ss;
  __syncthreads();
  for (int off = 128; off > 0; off >>= 1) {
    if (tid < off) s_red[tid] += s_red[tid + off];
    __syncthreads();
  }
  const float sc = rsqrtf(s_red[0] / (float)INTER_ + EPSF);
#pragma unroll
  for (int k = 0; k < 8; ++k) {
    const int i = tid + k * 256;
    out[(size_t)row * INTER_ + i] = f2bfu(nw[i] * yf[k] * sc);
  }
}

extern "C" void kernel_launch(void* const* d_in, const int* in_sizes, int n_in,
                              void* d_out, int out_size, void* d_ws,
                              size_t ws_size, hipStream_t stream) {
  const float* hid = (const float*)d_in[0];
  const float* ipw = (const float*)d_in[1];
  const float* cw = (const float*)d_in[2];
  const float* cb = (const float*)d_in[3];
  const float* dtb = (const float*)d_in[4];
  const float* alog = (const float*)d_in[5];
  const float* Dv = (const float*)d_in[6];
  const float* nw = (const float*)d_in[7];
  const float* opw = (const float*)d_in[8];
  float* out = (float*)d_out;

  char* ws = (char*)d_ws;
  size_t off = 0;
  auto alloc = [&](size_t bytes) -> void* {
    void* p = ws + off;
    off += (bytes + 255) & ~(size_t)255;
    return p;
  };
  const size_t MS = (size_t)B_ * S_;  // 8192 rows
  float* proj   = (float*)alloc(MS * PROJ_ * 4);
  float* hbuf   = (float*)alloc(MS * INTER_ * 4);  // also y (aliased)
  float* Bbuf   = (float*)alloc(MS * NST_ * 4);
  float* Cbuf   = (float*)alloc(MS * NST_ * 4);
  float* dtbuf  = (float*)alloc(MS * NH_ * 4);
  float* cAg    = (float*)alloc((size_t)B_ * NC_ * NH_ * CHK_ * 4);
  float* gws    = (float*)alloc((size_t)B_ * NH_ * NC_ * 4);
  float* states = (float*)alloc((size_t)B_ * NC_ * NH_ * HD_ * NST_ * 4);
  unsigned short* prevbf = (unsigned short*)alloc((size_t)B_ * NC_ * NH_ * HD_ * NST_ * 2);
  unsigned short* G      = (unsigned short*)alloc((size_t)B_ * NC_ * NH_ * CHK_ * CHK_ * 2);
  unsigned short* Abf    = (unsigned short*)alloc(MS * HSZ_ * 2);
  unsigned short* Wbfin  = (unsigned short*)alloc((size_t)4480 * HSZ_ * 2);  // padded to 35*128 rows
  unsigned short* ynorm  = (unsigned short*)alloc(MS * INTER_ * 2);
  unsigned short* Wbfout = (unsigned short*)alloc((size_t)HSZ_ * INTER_ * 2);
  (void)in_sizes; (void)n_in; (void)out_size; (void)ws_size;

  k_cvt_bf16<<<2048, 256, 0, stream>>>(hid, Abf, (long)MS * HSZ_);
  k_cvt_bf16<<<2048, 256, 0, stream>>>(ipw, Wbfin, (long)PROJ_ * HSZ_);
  k_cvt_bf16<<<1024, 256, 0, stream>>>(opw, Wbfout, (long)HSZ_ * INTER_);

  k_gemm_bf16<<<dim3(35, 64), 256, 0, stream>>>(Abf, Wbfin, proj,
                                                (int)MS, PROJ_, HSZ_);
  k_conv_act<<<(int)MS, 256, 0, stream>>>(proj, cw, cb, dtb, hbuf, Bbuf, Cbuf,
                                          dtbuf);
  k_chunk_state<<<B_ * NC_ * NH_, 256, 0, stream>>>(hbuf, Bbuf, dtbuf, alog,
                                                    cAg, gws, states);
  k_scan<<<B_ * NH_, 256, 0, stream>>>(states, gws, prevbf);
  k_chunk_G<<<B_ * NC_ * NH_ * 4, 256, 0, stream>>>(Bbuf, Cbuf, cAg, G);
  k_chunk_out<<<B_ * NC_ * NH_, 256, 0, stream>>>(Cbuf, G, prevbf, dtbuf, cAg,
                                                  Dv, hbuf);
  k_norm<<<(int)MS, 256, 0, stream>>>(hbuf, proj, nw, ynorm);
  k_gemm_bf16<<<dim3(8, 64), 256, 0, stream>>>(ynorm, Wbfout, out,
                                               (int)MS, HSZ_, INTER_);
}
